// MultiHeadSelfAttention_2_80590766342908
// MI455X (gfx1250) — matrix operand order repaired and checked by reading, not compiled, not run on hardware
//
#include <hip/hip_runtime.h>
#include <hip/hip_bf16.h>

// ---------------------------------------------------------------------------
// MultiHeadSelfAttention forward for MI455X (gfx1250), wave32 + WMMA f16.
//   B=8, S=1024, IN_DIM=D_MODEL=768, H=12, DK=64
// Pass 0: one-shot f32 -> f16 conversion of X, Wq, Wk, Wv
// Pass 1: fused QKV projection GEMM (f16 in, f32 WMMA accum, 32x64 tile/wave)
// Pass 2: attention, 32 queries/wave (scores WMMA -> exp/mask -> LDS relayout
//         -> context WMMA), K/V fragments reused across both query tiles.
// ---------------------------------------------------------------------------

typedef _Float16 half_t;
typedef __attribute__((ext_vector_type(16))) _Float16 v16h;
typedef __attribute__((ext_vector_type(8)))  _Float16 v8h;
typedef __attribute__((ext_vector_type(8)))  float    v8f;
typedef __attribute__((ext_vector_type(4)))  float    v4f;

#define NB   8
#define NS   1024
#define ND   768
#define NH   12
#define NDK  64
#define NTOK (NB * NS)                         // 8192
#define ELEM_PER_MAT ((size_t)NB * NS * ND)    // 6291456 halves = 12 MB
#define W_ELEMS      ((size_t)ND * ND)         // 589824 halves

// WMMA f16 16x16x32 fragment layouts (wave32), lane = g*16 + l:
//   A (16x32, MxK): row m=l; half i<8 -> K=g*8+i, i>=8 -> K=16+g*8+(i-8)
//   B (32x16, KxN): col n=l; half i -> same K as A half i (table order)
//   C (16x16 f32):  VGPR r -> (M=r+8*g, N=l)

static __device__ __forceinline__ v16h wmma_cat(v8h lo, v8h hi) {
    return __builtin_shufflevector(lo, hi, 0,1,2,3,4,5,6,7,8,9,10,11,12,13,14,15);
}

// Wave-local LDS fence: this wave's ds_stores complete before its ds_loads.
// (LDS ops are in-order per wave; we only need the counter wait + a compiler
//  scheduling barrier -- no workgroup barrier since staging is wave-private.)
static __device__ __forceinline__ void lds_fence_wave() {
    asm volatile("s_wait_dscnt 0x0" ::: "memory");
}
static __device__ __forceinline__ void compiler_fence() {
    asm volatile("" ::: "memory");
}

// ---------------------------------------------------------------------------
// Pass 0: elementwise f32 -> f16, 8 elements / thread (n must be multiple of 8)
// ---------------------------------------------------------------------------
__global__ __launch_bounds__(256) void cvt_f32_to_f16_kernel(
    const float* __restrict__ src, half_t* __restrict__ dst, int n)
{
    int i = (blockIdx.x * 256 + threadIdx.x) * 8;
    if (i >= n) return;
    v4f a = *(const v4f*)(src + i);
    v4f b = *(const v4f*)(src + i + 4);
    v8h o;
#pragma unroll
    for (int j = 0; j < 4; ++j) {
        o[j]     = (half_t)a[j];
        o[4 + j] = (half_t)b[j];
    }
    *(v8h*)(dst + i) = o;
}

// ---------------------------------------------------------------------------
// Pass 1: C[m,n] = sum_k Xh[m,k] * Wh[n,k] + bias[n]
//   MAT 0 -> q (scaled by 1/8, layout [B,H,S,DK])
//   MAT 1 -> k (layout [B,H,S,DK])
//   MAT 2 -> v (layout [B,H,DK,S]  == transposed for stage-2 B fragments)
// One wave per 32x64 output tile: 2 A frags x 4 B frags -> 8 WMMAs / k-step.
// ---------------------------------------------------------------------------
template <int MAT>
__global__ __launch_bounds__(256) void qkv_proj_kernel(
    const half_t* __restrict__ Xh, const half_t* __restrict__ Wh,
    const float* __restrict__ bias, half_t* __restrict__ dst)
{
    const int lane = threadIdx.x & 31;
    const int g    = lane >> 4;
    const int l    = lane & 15;

    const int w  = blockIdx.x * 8 + (threadIdx.x >> 5);  // 0..3071
    const int m0 = (w / (ND / 64)) * 32;                 // 256 m-tiles of 32
    const int n0 = (w % (ND / 64)) * 64;                 // 12 n-tiles of 64

    const half_t* Arow0 = Xh + (size_t)(m0 + l) * ND;
    const half_t* Arow1 = Xh + (size_t)(m0 + 16 + l) * ND;
    const half_t* Brow[4] = {
        Wh + (size_t)(n0 + l) * ND,
        Wh + (size_t)(n0 + 16 + l) * ND,
        Wh + (size_t)(n0 + 32 + l) * ND,
        Wh + (size_t)(n0 + 48 + l) * ND,
    };

    v8f acc[2][4] = {};
    for (int kb = 0; kb < ND; kb += 32) {
        v16h a0 = wmma_cat(*(const v8h*)(Arow0 + kb + g * 8),
                           *(const v8h*)(Arow0 + kb + 16 + g * 8));
        v16h a1 = wmma_cat(*(const v8h*)(Arow1 + kb + g * 8),
                           *(const v8h*)(Arow1 + kb + 16 + g * 8));
#pragma unroll
        for (int j = 0; j < 4; ++j) {
            v16h bfrag = wmma_cat(*(const v8h*)(Brow[j] + kb + g * 8),
                                  *(const v8h*)(Brow[j] + kb + 16 + g * 8));
            acc[0][j] = __builtin_amdgcn_wmma_f32_16x16x32_f16(false, a0, false, bfrag,
                                                               (short)0, acc[0][j], false, false);
            acc[1][j] = __builtin_amdgcn_wmma_f32_16x16x32_f16(false, a1, false, bfrag,
                                                               (short)0, acc[1][j], false, false);
        }
    }

    // store the eight 16x16 C fragments with bias (+ 1/sqrt(DK) folded into q)
#pragma unroll
    for (int j = 0; j < 4; ++j) {
        const int   nglob = n0 + j * 16 + l;
        const float bval  = bias[nglob];
        const int   h     = nglob >> 6;     // / DK
        const int   dk    = nglob & 63;
#pragma unroll
        for (int t = 0; t < 2; ++t) {
            if (MAT == 2) {
                // [B,H,DK,S]: the 8 rows are consecutive s -> one b128 store
                v8h o;
#pragma unroll
                for (int r = 0; r < 8; ++r)
                    o[r] = (half_t)(acc[t][j][r] + bval);
                int mglob = m0 + t * 16 + 8 * g;     // r = 0 row
                int bidx  = mglob >> 10;
                int s     = mglob & 1023;
                size_t idx = (((size_t)(bidx * NH + h)) * NDK + dk) * NS + s;
                *(v8h*)(dst + idx) = o;
            } else {
#pragma unroll
                for (int r = 0; r < 8; ++r) {
                    int   mglob = m0 + t * 16 + r + 8 * g;
                    int   bidx  = mglob >> 10;       // / S
                    int   s     = mglob & 1023;
                    float v     = acc[t][j][r] + bval;
                    if (MAT == 0) v *= 0.125f;       // 1/sqrt(64)
                    size_t idx = (((size_t)(bidx * NH + h)) * NS + s) * NDK + dk;
                    dst[idx] = (half_t)v;
                }
            }
        }
    }
}

// ---------------------------------------------------------------------------
// Pass 2: attention. 4 waves / block; each wave owns 32 queries of one (b,h).
// K/V fragments loaded once per 32-key block, reused by both query tiles.
// ---------------------------------------------------------------------------
__global__ __launch_bounds__(128) void attn_kernel(
    const half_t* __restrict__ qf, const half_t* __restrict__ kf,
    const half_t* __restrict__ vfT, const int* __restrict__ length,
    float* __restrict__ out)
{
    __shared__ __attribute__((aligned(32))) half_t sc[4][2][16 * 32]; // staging / wave / q-tile

    const int lane = threadIdx.x & 31;
    const int wid  = threadIdx.x >> 5;
    const int g    = lane >> 4;
    const int l    = lane & 15;

    const int blk  = blockIdx.x;          // B*H*(S/128) = 768
    const int qgrp = blk & 7;             // S/128 = 8
    const int bh   = blk >> 3;            // 0..95
    const int b    = bh / NH;
    const int h    = bh % NH;
    const int q0   = qgrp * 128 + wid * 32;
    const int len  = length[b];

    const half_t* qbase = qf  + (size_t)bh * NS * NDK;
    const half_t* kbase = kf  + (size_t)bh * NS * NDK;
    const half_t* vbase = vfT + (size_t)bh * NDK * NS;

    // q A-fragments for both query tiles (held across key loop), DK split 0/32
    v16h qa[2][2];
#pragma unroll
    for (int t = 0; t < 2; ++t) {
        const half_t* qrow = qbase + (size_t)(q0 + t * 16 + l) * NDK;
        qa[t][0] = *(const v16h*)(qrow + g * 16);
        qa[t][1] = *(const v16h*)(qrow + 32 + g * 16);
    }

    v8f   ctx[2][4] = {};
    float psum[2][8] = {};

    for (int kb = 0; kb < NS; kb += 32) {
        // software-prefetch next key block's K and V rows (global_prefetch_b8)
        if (kb + 32 < NS) {
            __builtin_prefetch(kbase + (size_t)(kb + 32 + l) * NDK + g * 16, 0, 0);
            __builtin_prefetch(kbase + (size_t)(kb + 48 + l) * NDK + g * 16, 0, 0);
            __builtin_prefetch(vbase + (size_t)l * NS + kb + 32 + g * 16, 0, 0);
            __builtin_prefetch(vbase + (size_t)(32 + l) * NS + kb + 32 + g * 16, 0, 0);
        }

        // ---- K fragments (shared by both query tiles) ----
        const half_t* krow0 = kbase + (size_t)(kb + l) * NDK;        // keys kb..kb+15
        const half_t* krow1 = kbase + (size_t)(kb + 16 + l) * NDK;   // keys kb+16..kb+31
        v16h kf00 = *(const v16h*)(krow0 + g * 16);
        v16h kf01 = *(const v16h*)(krow0 + 32 + g * 16);
        v16h kf10 = *(const v16h*)(krow1 + g * 16);
        v16h kf11 = *(const v16h*)(krow1 + 32 + g * 16);

        // ---- scores for both q tiles: two 16x16 key-half tiles each ----
        v8f s0[2], s1[2];
#pragma unroll
        for (int t = 0; t < 2; ++t) {
            v8f z0 = {}, z1 = {};
            z0 = __builtin_amdgcn_wmma_f32_16x16x32_f16(false, qa[t][0], false, kf00, (short)0, z0, false, false);
            z0 = __builtin_amdgcn_wmma_f32_16x16x32_f16(false, qa[t][1], false, kf01, (short)0, z0, false, false);
            z1 = __builtin_amdgcn_wmma_f32_16x16x32_f16(false, qa[t][0], false, kf10, (short)0, z1, false, false);
            z1 = __builtin_amdgcn_wmma_f32_16x16x32_f16(false, qa[t][1], false, kf11, (short)0, z1, false, false);
            s0[t] = z0;
            s1[t] = z1;
        }

        // ---- exp + key-length mask, stage to LDS in A-fragment source layout ----
        const bool val0 = (kb + l) < len;
        const bool val1 = (kb + 16 + l) < len;
        compiler_fence();   // keep prior iteration's LDS reads before these writes
#pragma unroll
        for (int t = 0; t < 2; ++t) {
            half_t* mysc = sc[wid][t];
#pragma unroll
            for (int r = 0; r < 8; ++r) {
                int   M  = r + 8 * g;
                float e0 = val0 ? __expf(s0[t][r]) : 0.f;
                float e1 = val1 ? __expf(s1[t][r]) : 0.f;
                psum[t][r] += e0 + e1;
                mysc[M * 32 + l]      = (half_t)e0;
                mysc[M * 32 + 16 + l] = (half_t)e1;
            }
        }
        lds_fence_wave();   // wave-private staging: ds waits only, no s_barrier

        // ---- V fragments (shared by both query tiles) ----
        v16h vfr[4];
#pragma unroll
        for (int j = 0; j < 4; ++j)
            vfr[j] = *(const v16h*)(vbase + (size_t)(j * 16 + l) * NS + kb + g * 16);

        // ---- context: ctx[t][:, j*16:+16] += attn_t(16x32) x V(32x16) ----
#pragma unroll
        for (int t = 0; t < 2; ++t) {
            const half_t* srow = sc[wid][t] + l * 32;
            v16h af = *(const v16h*)(srow + g * 16);
#pragma unroll
            for (int j = 0; j < 4; ++j)
                ctx[t][j] = __builtin_amdgcn_wmma_f32_16x16x32_f16(false, af, false, vfr[j],
                                                                   (short)0, ctx[t][j], false, false);
        }
        compiler_fence();
    }

    // ---- reduce row sums within each 16-lane group (same M set) ----
#pragma unroll
    for (int t = 0; t < 2; ++t)
#pragma unroll
        for (int r = 0; r < 8; ++r)
#pragma unroll
            for (int off = 1; off < 16; off <<= 1)
                psum[t][r] += __shfl_xor(psum[t][r], off, 32);

    // ---- normalize and store [B, S, H*DK] f32 ----
#pragma unroll
    for (int t = 0; t < 2; ++t) {
#pragma unroll
        for (int r = 0; r < 8; ++r) {
            float inv = 1.0f / (psum[t][r] + 1e-8f);
            int   s   = q0 + t * 16 + r + 8 * g;
            float* orow = out + ((size_t)(b * NS + s)) * ND + h * NDK;
            orow[l]      = ctx[t][0][r] * inv;
            orow[16 + l] = ctx[t][1][r] * inv;
            orow[32 + l] = ctx[t][2][r] * inv;
            orow[48 + l] = ctx[t][3][r] * inv;
        }
    }
}

// ---------------------------------------------------------------------------
extern "C" void kernel_launch(void* const* d_in, const int* in_sizes, int n_in,
                              void* d_out, int out_size, void* d_ws, size_t ws_size,
                              hipStream_t stream) {
    const float* Q      = (const float*)d_in[0];
    const int*   length = (const int*)d_in[1];
    const float* Wq     = (const float*)d_in[2];
    const float* bq     = (const float*)d_in[3];
    const float* Wk     = (const float*)d_in[4];
    const float* bk     = (const float*)d_in[5];
    const float* Wv     = (const float*)d_in[6];
    const float* bv     = (const float*)d_in[7];

    // workspace layout (halves): qf | kf | vf | Xh | Wqh | Wkh | Wvh  (~52 MB)
    half_t* qf  = (half_t*)d_ws;
    half_t* kf  = qf + ELEM_PER_MAT;
    half_t* vf  = kf + ELEM_PER_MAT;
    half_t* Xh  = vf + ELEM_PER_MAT;
    half_t* Wqh = Xh + ELEM_PER_MAT;
    half_t* Wkh = Wqh + W_ELEMS;
    half_t* Wvh = Wkh + W_ELEMS;

    // Pass 0: f32 -> f16 conversions (amortized once; removes cvt from GEMM loop)
    cvt_f32_to_f16_kernel<<<(int)(ELEM_PER_MAT / 8 + 255) / 256, 256, 0, stream>>>(Q, Xh, (int)ELEM_PER_MAT);
    cvt_f32_to_f16_kernel<<<(int)(W_ELEMS / 8 + 255) / 256, 256, 0, stream>>>(Wq, Wqh, (int)W_ELEMS);
    cvt_f32_to_f16_kernel<<<(int)(W_ELEMS / 8 + 255) / 256, 256, 0, stream>>>(Wk, Wkh, (int)W_ELEMS);
    cvt_f32_to_f16_kernel<<<(int)(W_ELEMS / 8 + 255) / 256, 256, 0, stream>>>(Wv, Wvh, (int)W_ELEMS);

    // Pass 1: 256 m-tiles x 12 n-tiles = 3072 waves per mat, 8 waves/block
    qkv_proj_kernel<0><<<384, 256, 0, stream>>>(Xh, Wqh, bq, qf);
    qkv_proj_kernel<1><<<384, 256, 0, stream>>>(Xh, Wkh, bk, kf);
    qkv_proj_kernel<2><<<384, 256, 0, stream>>>(Xh, Wvh, bv, vf);

    // Pass 2: B*H*(S/128) = 768 blocks of 4 waves (32 queries per wave)
    attn_kernel<<<768, 128, 0, stream>>>(qf, kf, vf, length, (float*)d_out);
}
